// SubspaceNet_14499809591568
// MI455X (gfx1250) — matrix operand order repaired and checked by reading, not compiled, not run on hardware
//
#include <hip/hip_runtime.h>

typedef _Float16 half_t;
typedef __attribute__((ext_vector_type(16))) _Float16 v16h;
typedef __attribute__((ext_vector_type(8)))  _Float16 v8h;
typedef __attribute__((ext_vector_type(2)))  _Float16 v2h;
typedef __attribute__((ext_vector_type(8)))  float    v8f;
typedef __attribute__((ext_vector_type(2)))  float    v2f;
typedef __attribute__((ext_vector_type(4)))  unsigned int u32x4;
typedef __attribute__((ext_vector_type(8)))  int      i32x8;
typedef __attribute__((ext_vector_type(4)))  int      i32x4;

// LDS layout (halves): X[32768] | Y[59520] | W0[16384] | W1[16384] | 64 f32 bias
#define X_HALVES 32768
#define Y_HALVES 59520
#define W_HALVES 16384
#define LDS_W0_BYTES ((X_HALVES + Y_HALVES) * 2)
#define LDS_W1_BYTES (LDS_W0_BYTES + W_HALVES * 2)
#define SMEM_BYTES (LDS_W1_BYTES + W_HALVES * 2 + 64 * 4)

// prepared-weight offsets in d_ws (halves), after the B*512 f32 Rx buffer
#define WP0 0
#define WP1 (WP0 + 16 * 32)
#define WP2 (WP1 + 32 * 128)
#define WP3 (WP2 + 64 * 256)
#define WP4 (WP3 + 32 * 512)
#define WP5 (WP4 + 16 * 256)

#if defined(__has_builtin)
#  if __has_builtin(__builtin_amdgcn_tensor_load_to_lds)
#    define HAVE_TDM 1
#  endif
#endif
#ifndef HAVE_TDM
#  define HAVE_TDM 0
#endif

// ---------------------------------------------------------------------------
// 1-D TDM descriptor: contiguous `count` elements (DS: 1=2B, 2=4B) from
// global -> LDS byte offset. Issued once per wave; completion via TENSORcnt.
// ---------------------------------------------------------------------------
#if HAVE_TDM
template<int DS>
__device__ __forceinline__ void tdm_load_1d(unsigned lds_byte, const void* gp,
                                            unsigned count) {
  const unsigned long long ga = (unsigned long long)(uintptr_t)gp;
  u32x4 g0 = {};
  g0[0] = 1u;                                            // count=1, user desc
  g0[1] = lds_byte;                                      // lds_addr
  g0[2] = (unsigned)ga;                                  // global_addr[31:0]
  g0[3] = (unsigned)((ga >> 32) & 0x01FFFFFFu) | (2u << 30); // addr[56:32]|type=2
  i32x8 g1 = {};
  g1[0] = (DS << 16);                                    // data_size
  g1[1] = (int)((count & 0xFFFFu) << 16);                // tensor_dim0[15:0]
  g1[2] = (int)(((count >> 16) & 0xFFFFu) | (1u << 16)); // dim0 hi | tensor_dim1=1
  g1[3] = (int)((count & 0xFFFFu) << 16);                // tile_dim0
  g1[4] = 1;                                             // tile_dim1=1
  g1[5] = (int)count;                                    // tensor_dim0_stride
  i32x4 z4 = {};
#if __clang_major__ >= 23
  i32x8 z8 = {};
  __builtin_amdgcn_tensor_load_to_lds(g0, g1, z4, z4, z8, 0);
#else
  __builtin_amdgcn_tensor_load_to_lds(g0, g1, z4, z4, 0);
#endif
}
#endif

// ---------------------------------------------------------------------------
// Weight prep: transform conv/deconv weights once into contiguous f16
// [co][K] tiles (K = (ky*2+kx)*CIN + ci_mem), zero-padded to 16 rows.
// ANTI_IN: the layer input is an anti-rectifier output stored interleaved
// (mem 2c = relu(x_c), 2c+1 = relu(-x_c)) -> permute ci accordingly.
// ---------------------------------------------------------------------------
template<int CIN,int COUT,int DECONV,int ANTI_IN>
__device__ __forceinline__ void prep_w(const float* __restrict__ wg,
                                       half_t* __restrict__ dst) {
  constexpr int KT = CIN * 4;
  constexpr int CO_R = ((COUT + 15) >> 4) * 16;
  for (int idx = threadIdx.x; idx < CO_R * KT; idx += 256) {
    const int co = idx / KT, k = idx - co * KT;
    float w = 0.0f;
    if (co < COUT) {
      const int kpos = k / CIN, ci_mem = k - kpos * CIN;
      int ci = ci_mem;
      if constexpr (ANTI_IN) ci = (ci_mem >> 1) + (ci_mem & 1) * (CIN / 2);
      const int ky = kpos >> 1, kx = kpos & 1;
      if constexpr (DECONV) w = wg[((ci * COUT + co) * 2 + (1 - ky)) * 2 + (1 - kx)];
      else                  w = wg[((co * CIN + ci) * 2 + ky) * 2 + kx];
    }
    dst[idx] = (half_t)w;
  }
}

__global__ __launch_bounds__(256)
void prep_kernel(const float* __restrict__ w1,  const float* __restrict__ w2,
                 const float* __restrict__ w3,  const float* __restrict__ dw2,
                 const float* __restrict__ dw3, const float* __restrict__ dw4,
                 half_t* __restrict__ wp) {
  switch (blockIdx.x) {
    case 0: prep_w<  8, 16, 0, 0>(w1,  wp + WP0); break;
    case 1: prep_w< 32, 32, 0, 1>(w2,  wp + WP1); break;
    case 2: prep_w< 64, 64, 0, 1>(w3,  wp + WP2); break;
    case 3: prep_w<128, 32, 1, 1>(dw2, wp + WP3); break;
    case 4: prep_w< 64, 16, 1, 1>(dw3, wp + WP4); break;
    case 5: prep_w< 32,  1, 1, 1>(dw4, wp + WP5); break;
  }
}

// ---------------------------------------------------------------------------
// One fused conv/deconv + bias + anti-rectifier step, all in LDS, WMMA f16.
// Activations: padded NHWC f16, (H+2)x(W+2)xC, zero border, anti interleaved.
// Weights for THIS layer were TDM-prefetched into wcur by the previous layer
// (or the preamble); this step prefetches wnext for the following layer.
// ---------------------------------------------------------------------------
template<int CIN,int COUT,int IH,int IW,int PAD,int ANTI,int GOUT>
__device__ __forceinline__
void layer_step(const half_t* __restrict__ in, half_t* __restrict__ out,
                half_t* __restrict__ wcur, const half_t* __restrict__ wsrc_cur,
                const half_t* __restrict__ wnext_src, unsigned wnext_byte,
                int wnext_count, float* __restrict__ blds,
                const float* __restrict__ bg, float* __restrict__ gout,
                const int tid)
{
  constexpr int OH = PAD ? IH + 1 : IH - 1;
  constexpr int OW = PAD ? IW + 1 : IW - 1;
  constexpr int KT = CIN * 4;
  constexpr int M  = OH * OW;
  constexpr int MT = (M + 15) >> 4;
  constexpr int NT = (COUT + 15) >> 4;
  constexpr int CO2 = ANTI ? 2 * COUT : COUT;
  constexpr int ADJ = PAD ? 0 : 1;
  static_assert(CIN % 8 == 0, "CIN must be a multiple of 8");

#if !HAVE_TDM
  {
    constexpr int CO_R = NT * 16;
    for (int idx = tid; idx < (CO_R * KT) / 8; idx += 256)
      ((v8h*)wcur)[idx] = ((const v8h*)wsrc_cur)[idx];
  }
#endif
  for (int c = tid; c < COUT; c += 256) blds[c] = bg[c];
  if constexpr (!GOUT) {
    for (int idx = tid; idx < ((OH + 2) * (OW + 2) * CO2) / 8; idx += 256)
      ((v8h*)out)[idx] = (v8h)0;
  }
#if HAVE_TDM
  if (tid < 32) __builtin_amdgcn_s_wait_tensorcnt(0);  // this layer's weights
#endif
  __syncthreads();
#if HAVE_TDM
  if (tid < 32 && wnext_src != nullptr)                // overlap next DMA
    tdm_load_1d<1>(wnext_byte, wnext_src, (unsigned)wnext_count);
#endif

  // ---- implicit GEMM: M = output pixels, N = Cout, K = Cin*4 ----
  const int lane = tid & 31;
  const int lr = lane & 15, hi = lane >> 4;
  for (int job = (tid >> 5); job < MT * NT; job += 8) {
    const int tm = job / NT, tn = job - tm * NT;
    int m = tm * 16 + lr; if (m > M - 1) m = M - 1;   // clamp (EXEC stays full)
    const int oy = m / OW, ox = m - oy * OW;
    const half_t* inH = &in[(oy * (IW + 2) + ox) * CIN + hi * 8];
    const half_t* wb  = &wcur[(tn * 16 + lr) * KT + hi * 8];
    v8f acc = {};
    #pragma unroll
    for (int kk = 0; kk < KT; kk += 32) {
      v16h a;
      #pragma unroll
      for (int run = 0; run < 2; ++run) {
        v8h v;
        if constexpr (CIN >= 32) {
          const int rb = kk + run * 16;               // folds to constants
          const int kpos = rb / CIN, ci0 = rb - kpos * CIN;
          const int off = (((kpos >> 1) + ADJ) * (IW + 2) + (kpos & 1) + ADJ) * CIN + ci0;
          v = *(const v8h*)(inH + off);
        } else {
          const int rr = kk + hi * 8 + run * 16;
          const int kpos = rr / CIN, ci0 = rr - kpos * CIN;
          const int off = (((kpos >> 1) + ADJ) * (IW + 2) + (kpos & 1) + ADJ) * CIN + ci0;
          v = *(const v8h*)(in + (oy * (IW + 2) + ox) * CIN + off);
        }
        #pragma unroll
        for (int e = 0; e < 8; ++e) a[run * 8 + e] = v[e];
      }
      const v8h b0 = *(const v8h*)(wb + kk), b1 = *(const v8h*)(wb + kk + 16);
      v16h bf;
      #pragma unroll
      for (int e = 0; e < 8; ++e) { bf[e] = b0[e]; bf[8 + e] = b1[e]; }
      acc = __builtin_amdgcn_wmma_f32_16x16x32_f16(false, a, false, bf,
                                                   (short)0, acc, false, false);
    }

    // ---- bias + interleaved anti store (one b32 per element pair) ----
    const int co = tn * 16 + lr;
    const float bv = GOUT ? blds[0] : blds[co];
    int mm = tm * 16 + hi * 8;
    int oy2 = mm / OW, ox2 = mm - oy2 * OW;
    int pix = (oy2 + 1) * (OW + 2) + (ox2 + 1);
    if (tm * 16 + 16 <= M) {                          // full tile: no guards
      #pragma unroll
      for (int r = 0; r < 8; ++r) {
        const float y = acc[r] + bv;
        if constexpr (GOUT) {
          if (lr == 0) gout[mm] = y;
        } else if constexpr (ANTI) {
          v2h pk; pk[0] = (half_t)fmaxf(y, 0.0f); pk[1] = (half_t)fmaxf(-y, 0.0f);
          *(v2h*)&out[pix * CO2 + 2 * co] = pk;
        } else {
          out[pix * CO2 + co] = (half_t)y;
        }
        ++mm; ++pix; if (++ox2 == OW) { ox2 = 0; pix += 2; }
      }
    } else {                                          // uniform tail tile
      #pragma unroll
      for (int r = 0; r < 8; ++r) {
        if (mm < M) {
          const float y = acc[r] + bv;
          if constexpr (GOUT) {
            if (lr == 0) gout[mm] = y;
          } else if constexpr (ANTI) {
            v2h pk; pk[0] = (half_t)fmaxf(y, 0.0f); pk[1] = (half_t)fmaxf(-y, 0.0f);
            *(v2h*)&out[pix * CO2 + 2 * co] = pk;
          } else {
            out[pix * CO2 + co] = (half_t)y;
          }
        }
        ++mm; ++pix; if (++ox2 == OW) { ox2 = 0; pix += 2; }
      }
    }
  }
  __syncthreads();
}

// ---------------------------------------------------------------------------
// Kernel 1: entire CNN per batch element fused in LDS (one WG per sample).
// ---------------------------------------------------------------------------
__global__ __launch_bounds__(256)
void subspace_cnn_kernel(const float* __restrict__ Rx_tau,
    const half_t* __restrict__ wp,
    const float* __restrict__ b1,  const float* __restrict__ b2,
    const float* __restrict__ b3,  const float* __restrict__ db2,
    const float* __restrict__ db3, const float* __restrict__ db4,
    float* __restrict__ RxOut)
{
  extern __shared__ char smem[];
  half_t* X  = (half_t*)smem;
  half_t* Y  = X + X_HALVES;
  half_t* W0 = Y + Y_HALVES;
  half_t* W1 = W0 + W_HALVES;
  float* BIAS = (float*)(W1 + W_HALVES);
  const int b = blockIdx.x, tid = threadIdx.x;
  const float* src = Rx_tau + (size_t)b * (8 * 32 * 16);

  __builtin_prefetch(wp, 0, 0);

  // preamble: DMA raw f32 sample -> W1 staging, layer-0 weights -> W0
#if HAVE_TDM
  if (tid < 32) {
    tdm_load_1d<2>(LDS_W1_BYTES, src, 4096);
    tdm_load_1d<1>(LDS_W0_BYTES, wp + WP0, 16 * 32);
  }
#endif
  for (int i = tid; i < (34 * 18 * 8) / 8; i += 256) ((v8h*)X)[i] = (v8h)0;
#if HAVE_TDM
  if (tid < 32) __builtin_amdgcn_s_wait_tensorcnt(0);
  __syncthreads();
  const float* stage = (const float*)(smem + LDS_W1_BYTES);
  for (int i = tid; i < 8 * 32 * 16; i += 256) {
    const int t = i >> 9, rem = i & 511, y = rem >> 4, x = rem & 15;
    X[((y + 1) * 18 + (x + 1)) * 8 + t] = (half_t)stage[i];
  }
#else
  __syncthreads();
  for (int i = tid; i < 8 * 32 * 16; i += 256) {
    const int t = i >> 9, rem = i & 511, y = rem >> 4, x = rem & 15;
    X[((y + 1) * 18 + (x + 1)) * 8 + t] = (half_t)src[i];
  }
#endif
  __syncthreads();

  layer_step<  8, 16, 32, 16, 0, 1, 0>(X, Y, W0, wp + WP0,
      wp + WP1, LDS_W1_BYTES, 32 * 128, BIAS, b1,  nullptr, tid);
  layer_step< 32, 32, 31, 15, 0, 1, 0>(Y, X, W1, wp + WP1,
      wp + WP2, LDS_W0_BYTES, 64 * 256, BIAS, b2,  nullptr, tid);
  layer_step< 64, 64, 30, 14, 0, 1, 0>(X, Y, W0, wp + WP2,
      wp + WP3, LDS_W1_BYTES, 32 * 512, BIAS, b3,  nullptr, tid);
  layer_step<128, 32, 29, 13, 1, 1, 0>(Y, X, W1, wp + WP3,
      wp + WP4, LDS_W0_BYTES, 16 * 256, BIAS, db2, nullptr, tid);
  layer_step< 64, 16, 30, 14, 1, 1, 0>(X, Y, W0, wp + WP4,
      wp + WP5, LDS_W1_BYTES, 16 * 128, BIAS, db3, nullptr, tid);
  layer_step< 32,  1, 31, 15, 1, 0, 1>(Y, nullptr, W1, wp + WP5,
      nullptr, 0, 0, BIAS, db4, RxOut + (size_t)b * 512, tid);
}

// ---------------------------------------------------------------------------
// Complex helpers for ESPRIT tail
// ---------------------------------------------------------------------------
struct cpx { float r, i; };
__device__ __forceinline__ cpx cmk(float r, float i){ return {r, i}; }
__device__ __forceinline__ cpx operator+(cpx a, cpx b){ return {a.r+b.r, a.i+b.i}; }
__device__ __forceinline__ cpx operator-(cpx a, cpx b){ return {a.r-b.r, a.i-b.i}; }
__device__ __forceinline__ cpx operator*(cpx a, cpx b){ return {a.r*b.r-a.i*b.i, a.r*b.i+a.i*b.r}; }
__device__ __forceinline__ cpx cconj(cpx a){ return {a.r, -a.i}; }
__device__ __forceinline__ cpx cscale(cpx a, float s){ return {a.r*s, a.i*s}; }
__device__ __forceinline__ float cabs2(cpx a){ return a.r*a.r + a.i*a.i; }
__device__ __forceinline__ cpx cdivc(cpx a, cpx b){
  const float d = b.r*b.r + b.i*b.i + 1e-30f;
  return {(a.r*b.r + a.i*b.i)/d, (a.i*b.r - a.r*b.i)/d};
}
__device__ __forceinline__ cpx csqrtc(cpx z){
  const float m = sqrtf(hypotf(z.r, z.i));
  const float t = 0.5f * atan2f(z.i, z.r);
  return {m * cosf(t), m * sinf(t)};
}
__device__ __forceinline__ cpx ccbrtc(cpx z){
  const float m = cbrtf(hypotf(z.r, z.i));
  const float t = atan2f(z.i, z.r) * (1.0f/3.0f);
  return {m * cosf(t), m * sinf(t)};
}

// ---------------------------------------------------------------------------
// Kernel 2: one wave per sample. Rz = K^H K + I via f32 WMMA 16x16x4,
// then wave-parallel complex Jacobi eigh, then lane-0 ESPRIT -> doa.
// ---------------------------------------------------------------------------
__global__ __launch_bounds__(32)
void subspace_esprit_kernel(const float* __restrict__ RxIn,
                            float* __restrict__ doa_out,
                            float* __restrict__ rz_out)
{
  __shared__ float Kr[256], Ki[256], Ar[256], Ai[256], Vr[256], Vi[256];
  const int b = blockIdx.x, lane = threadIdx.x;
  const int lr = lane & 15, hi = lane >> 4;
  const float* rx = RxIn + (size_t)b * 512;

  for (int i = lane; i < 256; i += 32) { Kr[i] = rx[i]; Ki[i] = rx[256 + i]; }
  for (int i = lane; i < 256; i += 32) { Vr[i] = ((i % 17) == 0) ? 1.0f : 0.0f; Vi[i] = 0.0f; }
  __syncthreads();

  // Gram products: Re = Kr^T Kr + Ki^T Ki ; Im = Kr^T Ki - Ki^T Kr
  v8f aR = {}, aP = {}, aN = {};
  #pragma unroll
  for (int kk = 0; kk < 16; kk += 4) {
    v2f fr, fi;
    #pragma unroll
    for (int r = 0; r < 2; ++r) {
      const int i = kk + hi * 2 + r;
      fr[r] = Kr[i * 16 + lr];
      fi[r] = Ki[i * 16 + lr];
    }
    aR = __builtin_amdgcn_wmma_f32_16x16x4_f32(false, fr, false, fr, (short)0, aR, false, false);
    aR = __builtin_amdgcn_wmma_f32_16x16x4_f32(false, fi, false, fi, (short)0, aR, false, false);
    aP = __builtin_amdgcn_wmma_f32_16x16x4_f32(false, fr, false, fi, (short)0, aP, false, false);
    aN = __builtin_amdgcn_wmma_f32_16x16x4_f32(false, fi, false, fr, (short)0, aN, false, false);
  }
  __syncthreads();
  float* rzb = rz_out + (size_t)b * 512;
  #pragma unroll
  for (int r = 0; r < 8; ++r) {
    const int m = r + hi * 8, n = lr;
    const float re = aR[r] + ((m == n) ? 1.0f : 0.0f);   // EPS = 1
    const float im = aP[r] - aN[r];
    Ar[m * 16 + n] = re; Ai[m * 16 + n] = im;
    rzb[(m * 16 + n) * 2 + 0] = re;
    rzb[(m * 16 + n) * 2 + 1] = im;
  }
  __syncthreads();

  // Cyclic complex Jacobi (Hermitian), 10 sweeps
  for (int sweep = 0; sweep < 10; ++sweep) {
    for (int p = 0; p < 15; ++p) {
      for (int q = p + 1; q < 16; ++q) {
        const float ar = Ar[p * 16 + q], ai = Ai[p * 16 + q];
        const float app = Ar[p * 16 + p], aqq = Ar[q * 16 + q];
        const float mag = sqrtf(ar * ar + ai * ai);
        float c = 1.0f, s = 0.0f, er = 1.0f, ei = 0.0f;
        if (mag > 1e-12f) {
          const float th = 0.5f * atan2f(2.0f * mag, app - aqq);
          c = cosf(th); s = sinf(th);
          er = ar / mag; ei = ai / mag;
        }
        __syncthreads();
        if (lane < 16) {            // rows p,q of A  (R^H A)
          const int j = lane;
          const float upr = Ar[p*16+j], upi = Ai[p*16+j];
          const float uqr = Ar[q*16+j], uqi = Ai[q*16+j];
          Ar[p*16+j] = c*upr - s*(er*uqr - ei*uqi);
          Ai[p*16+j] = c*upi - s*(er*uqi + ei*uqr);
          Ar[q*16+j] = s*(er*upr + ei*upi) + c*uqr;
          Ai[q*16+j] = s*(er*upi - ei*upr) + c*uqi;
        } else {                    // eigenvector columns p,q  (V R)
          const int j = lane - 16;
          const float vpr = Vr[j*16+p], vpi = Vi[j*16+p];
          const float vqr = Vr[j*16+q], vqi = Vi[j*16+q];
          Vr[j*16+p] = c*vpr - s*(er*vqr + ei*vqi);
          Vi[j*16+p] = c*vpi - s*(er*vqi - ei*vqr);
          Vr[j*16+q] = s*(er*vpr - ei*vpi) + c*vqr;
          Vi[j*16+q] = s*(er*vpi + ei*vpr) + c*vqi;
        }
        __syncthreads();
        if (lane < 16) {            // columns p,q of A  (A R)
          const int j = lane;
          const float upr = Ar[j*16+p], upi = Ai[j*16+p];
          const float uqr = Ar[j*16+q], uqi = Ai[j*16+q];
          Ar[j*16+p] = c*upr - s*(er*uqr + ei*uqi);
          Ai[j*16+p] = c*upi - s*(er*uqi - ei*uqr);
          Ar[j*16+q] = s*(er*upr - ei*upi) + c*uqr;
          Ai[j*16+q] = s*(er*upi + ei*upr) + c*uqi;
        }
        __syncthreads();
      }
    }
  }

  if (lane == 0) {
    // top-3 eigenvalues (descending)
    float ev[16];
    for (int i = 0; i < 16; ++i) ev[i] = Ar[i * 16 + i];
    int id[3];
    for (int m = 0; m < 3; ++m) {
      int bi = 0; float bv = -3.4e38f;
      for (int i = 0; i < 16; ++i) {
        bool used = false;
        for (int t = 0; t < m; ++t) used |= (id[t] == i);
        if (!used && ev[i] > bv) { bv = ev[i]; bi = i; }
      }
      id[m] = bi;
    }
    // ESPRIT: phi = (Us_up^H Us_up)^-1 (Us_up^H Us_lo), 3x3 complex
    cpx G[3][3], H[3][3];
    for (int a = 0; a < 3; ++a) for (int c2 = 0; c2 < 3; ++c2) { G[a][c2] = cmk(0,0); H[a][c2] = cmk(0,0); }
    for (int i = 0; i < 15; ++i) {
      cpx ua[3], ub[3];
      for (int m = 0; m < 3; ++m) {
        ua[m] = cmk(Vr[i * 16 + id[m]],       Vi[i * 16 + id[m]]);
        ub[m] = cmk(Vr[(i + 1) * 16 + id[m]], Vi[(i + 1) * 16 + id[m]]);
      }
      for (int a = 0; a < 3; ++a) for (int c2 = 0; c2 < 3; ++c2) {
        G[a][c2] = G[a][c2] + cconj(ua[a]) * ua[c2];
        H[a][c2] = H[a][c2] + cconj(ua[a]) * ub[c2];
      }
    }
    const cpx c00 = G[1][1]*G[2][2] - G[1][2]*G[2][1];
    const cpx c01 = G[1][2]*G[2][0] - G[1][0]*G[2][2];
    const cpx c02 = G[1][0]*G[2][1] - G[1][1]*G[2][0];
    const cpx det = G[0][0]*c00 + G[0][1]*c01 + G[0][2]*c02;
    cpx Gi[3][3];
    Gi[0][0] = cdivc(c00, det);
    Gi[0][1] = cdivc(G[0][2]*G[2][1] - G[0][1]*G[2][2], det);
    Gi[0][2] = cdivc(G[0][1]*G[1][2] - G[0][2]*G[1][1], det);
    Gi[1][0] = cdivc(c01, det);
    Gi[1][1] = cdivc(G[0][0]*G[2][2] - G[0][2]*G[2][0], det);
    Gi[1][2] = cdivc(G[0][2]*G[1][0] - G[0][0]*G[1][2], det);
    Gi[2][0] = cdivc(c02, det);
    Gi[2][1] = cdivc(G[0][1]*G[2][0] - G[0][0]*G[2][1], det);
    Gi[2][2] = cdivc(G[0][0]*G[1][1] - G[0][1]*G[1][0], det);
    cpx phi[3][3];
    for (int a = 0; a < 3; ++a) for (int c2 = 0; c2 < 3; ++c2) {
      cpx s = cmk(0,0);
      for (int k = 0; k < 3; ++k) s = s + Gi[a][k] * H[k][c2];
      phi[a][c2] = s;
    }
    const cpx tr  = phi[0][0] + phi[1][1] + phi[2][2];
    const cpx c1s = (phi[0][0]*phi[1][1] - phi[0][1]*phi[1][0])
                  + (phi[0][0]*phi[2][2] - phi[0][2]*phi[2][0])
                  + (phi[1][1]*phi[2][2] - phi[1][2]*phi[2][1]);
    const cpx dph = phi[0][0]*(phi[1][1]*phi[2][2] - phi[1][2]*phi[2][1])
                  - phi[0][1]*(phi[1][0]*phi[2][2] - phi[1][2]*phi[2][0])
                  + phi[0][2]*(phi[1][0]*phi[2][1] - phi[1][1]*phi[2][0]);
    const cpx A = cscale(tr, -1.0f), Bc = c1s, Cc = cscale(dph, -1.0f);
    const cpx a3 = cscale(A, 1.0f/3.0f);
    const cpx p  = Bc - cscale(A*A, 1.0f/3.0f);
    const cpx qq = cscale(A*A*A, 2.0f/27.0f) - cscale(A*Bc, 1.0f/3.0f) + Cc;
    const cpx hq = cscale(qq, 0.5f);
    const cpx disc = hq*hq + cscale(p*p*p, 1.0f/27.0f);
    const cpx sq = csqrtc(disc);
    cpx u3 = cscale(hq, -1.0f) + sq;
    if (cabs2(u3) < 1e-18f) u3 = cscale(hq, -1.0f) - sq;
    const cpx u = ccbrtc(u3);
    const cpx v = (cabs2(u) > 1e-30f) ? cscale(cdivc(p, u), -1.0f/3.0f) : cmk(0,0);
    const cpx w1r = cmk(-0.5f,  0.8660254037844386f);
    const cpx w2r = cmk(-0.5f, -0.8660254037844386f);
    cpx lam[3];
    lam[0] = u + v - a3;
    lam[1] = u*w1r + v*w2r - a3;
    lam[2] = u*w2r + v*w1r - a3;
    for (int k = 0; k < 3; ++k) {
      const float ang = atan2f(lam[k].i, lam[k].r);
      doa_out[b * 3 + k] = -asinf(ang * 0.31830988618379067f);
    }
  }
}

// ---------------------------------------------------------------------------
extern "C" void kernel_launch(void* const* d_in, const int* in_sizes, int n_in,
                              void* d_out, int out_size, void* d_ws, size_t ws_size,
                              hipStream_t stream) {
  const float* Rx_tau = (const float*)d_in[0];
  // d_in[1] is the python scalar M (== 3, baked in at compile time)
  const float* w1  = (const float*)d_in[2];
  const float* b1  = (const float*)d_in[3];
  const float* w2  = (const float*)d_in[4];
  const float* b2  = (const float*)d_in[5];
  const float* w3  = (const float*)d_in[6];
  const float* b3  = (const float*)d_in[7];
  const float* dw2 = (const float*)d_in[8];
  const float* db2 = (const float*)d_in[9];
  const float* dw3 = (const float*)d_in[10];
  const float* db3 = (const float*)d_in[11];
  const float* dw4 = (const float*)d_in[12];
  const float* db4 = (const float*)d_in[13];

  const int Bn = in_sizes[0] / (8 * 32 * 16);          // 2048
  float*  Rx = (float*)d_ws;                           // B x 32 x 16 f32
  half_t* wp = (half_t*)((char*)d_ws + (size_t)Bn * 512 * sizeof(float));
  float* out = (float*)d_out;
  float* doa = out;                                    // B*3 floats
  float* rz  = out + (size_t)Bn * 3;                   // B*16*16 complex (re,im)

  (void)hipFuncSetAttribute(reinterpret_cast<const void*>(subspace_cnn_kernel),
                            hipFuncAttributeMaxDynamicSharedMemorySize, SMEM_BYTES);

  prep_kernel<<<6, 256, 0, stream>>>(w1, w2, w3, dw2, dw3, dw4, wp);

  subspace_cnn_kernel<<<Bn, 256, SMEM_BYTES, stream>>>(
      Rx_tau, wp, b1, b2, b3, db2, db3, db4, Rx);

  subspace_esprit_kernel<<<Bn, 32, 0, stream>>>(Rx, doa, rz);
}
